// FairTokenMoE_11029476016328
// MI455X (gfx1250) — compile-verified, matrix operand order checked by reading, not run on hardware
//
#include <hip/hip_runtime.h>

// ---------------- problem constants ----------------
#define T_TOK   197
#define B_BAT   64
#define F_DIM   768
#define N_EXP   8
#define K_TOP   2
#define CAP     49
#define H_DIM   3072          // 4*F
#define G_DIM   192           // F/4
#define TB      (T_TOK * B_BAT)   // 12608 tokens
#define SLOTS   3200          // per-expert list padded to 128 (>= B*CAP = 3136)

// GEMM tiling: block 128x128x32, 4 waves, 64x64 wave tile, 4x4 WMMA subtiles
#define BM 128
#define BN 128
#define BK 32

typedef __attribute__((ext_vector_type(16))) __bf16          v16bf;
typedef __attribute__((ext_vector_type(8)))  float           v8f;
typedef __attribute__((ext_vector_type(8)))  unsigned short  u16x8;
typedef __attribute__((ext_vector_type(4)))  unsigned short  u16x4;
typedef __attribute__((ext_vector_type(4)))  unsigned int    u32x4;
typedef __attribute__((ext_vector_type(8)))  int             i32x8;
typedef __attribute__((ext_vector_type(4)))  int             i32x4;

union Frag { v16bf bf; u16x8 h[2]; };

__device__ __forceinline__ unsigned short f2bf(float f) {
    unsigned u = __float_as_uint(f);
    unsigned r = u + 0x7fffu + ((u >> 16) & 1u);   // round-to-nearest-even
    return (unsigned short)(r >> 16);
}

// byte offset of a generic pointer inside LDS (addrspacecast generic -> local)
__device__ __forceinline__ unsigned lds_offset(const void* p) {
    return (unsigned)(unsigned long long)(const __attribute__((address_space(3))) void*)p;
}

// ---- Tensor Data Mover: 2-D bf16 tile (tile_d1 rows x tile_d0 elems) -> LDS -----
// D# packing per CDNA5 ISA (group0 128b, group1 256b).
__device__ __forceinline__ void tdm_load_2d(
    unsigned lds_off, const void* gaddr,
    unsigned tensor_d0, unsigned tensor_d1,
    unsigned tile_d0, unsigned tile_d1, unsigned stride_d0)
{
    unsigned long long ga = (unsigned long long)gaddr;
    u32x4 g0;
    g0[0] = 1u;                                            // count=1 (valid), user mode
    g0[1] = lds_off;                                       // lds_addr
    g0[2] = (unsigned)(ga & 0xffffffffu);                  // global_addr[31:0]
    g0[3] = (unsigned)((ga >> 32) & 0x01ffffffu) | (2u << 30); // addr[56:32], type=2
    i32x8 g1;
    g1[0] = (int)(1u << 16);                               // data_size=1 (2B), mask=0
    g1[1] = (int)((tensor_d0 & 0xffffu) << 16);            // tensor_dim0[15:0] @ bits 63:48
    g1[2] = (int)(((tensor_d0 >> 16) & 0xffffu) | ((tensor_d1 & 0xffffu) << 16));
    g1[3] = (int)(((tensor_d1 >> 16) & 0xffffu) | ((tile_d0 & 0xffffu) << 16));
    g1[4] = (int)(tile_d1 & 0xffffu);                      // tile_dim1, tile_dim2=0
    g1[5] = (int)stride_d0;                                // tensor_dim0_stride[31:0]
    g1[6] = 0;                                             // stride hi, dim1_stride=0
    g1[7] = 0;
    i32x4 z4 = {0, 0, 0, 0};
    i32x8 z8 = {0, 0, 0, 0, 0, 0, 0, 0};
    __builtin_amdgcn_tensor_load_to_lds(g0, g1, z4, z4, z8, 0);
}

// load one 16x32 bf16 fragment; LDS layout is [row][k] with k-pitch = BK
__device__ __forceinline__ v16bf load_frag(const unsigned short* lds, int rc, int lane) {
    const int kb = (lane < 16) ? 0 : 8;
    Frag f;
    f.h[0] = *(const u16x8*)(lds + rc * BK + kb);
    f.h[1] = *(const u16x8*)(lds + rc * BK + kb + 16);
    return f.bf;
}

// 16 WMMAs of one 64x64 wave tile over a K=32 slab
__device__ __forceinline__ void mma_step(
    const unsigned short* As, const unsigned short* Bs,
    v8f acc[4][4], int wm, int wn, int lane)
{
    v16bf af[4], bf[4];
    #pragma unroll
    for (int i = 0; i < 4; ++i) af[i] = load_frag(As, wm * 64 + i * 16 + (lane & 15), lane);
    #pragma unroll
    for (int j = 0; j < 4; ++j) bf[j] = load_frag(Bs, wn * 64 + j * 16 + (lane & 15), lane);
    #pragma unroll
    for (int i = 0; i < 4; ++i)
        #pragma unroll
        for (int j = 0; j < 4; ++j)
            acc[i][j] = __builtin_amdgcn_wmma_f32_16x16x32_bf16(
                false, af[i], false, bf[j], (short)0, acc[i][j], false, false);
}

// ---------------- 1) gating ------------------------------------------------------
__global__ __launch_bounds__(256) void gate_kernel(
    const float* __restrict__ x,  const float* __restrict__ Wg1,
    const float* __restrict__ bg1,const float* __restrict__ Wg2,
    const float* __restrict__ bg2,float* __restrict__ cw)
{
    const int tok = blockIdx.x;
    const int tid = threadIdx.x;
    __shared__ float xs[F_DIM];
    __shared__ float hg[G_DIM];
    __shared__ float lg[N_EXP];

    for (int i = tid; i < F_DIM; i += 256) xs[i] = x[(size_t)tok * F_DIM + i];
    __syncthreads();
    if (tid < G_DIM) {
        float a = bg1[tid];
        #pragma unroll 4
        for (int k = 0; k < F_DIM; ++k) a += xs[k] * Wg1[k * G_DIM + tid];
        hg[tid] = fmaxf(a, 0.f);
    }
    __syncthreads();
    if (tid < N_EXP) {
        float a = bg2[tid];
        #pragma unroll 4
        for (int k = 0; k < G_DIM; ++k) a += hg[k] * Wg2[k * N_EXP + tid];
        lg[tid] = a;
    }
    __syncthreads();
    if (tid < N_EXP) {
        float m = lg[0];
        #pragma unroll
        for (int j = 1; j < N_EXP; ++j) m = fmaxf(m, lg[j]);
        float s = 0.f;
        #pragma unroll
        for (int j = 0; j < N_EXP; ++j) s += expf(lg[j] - m);
        const float g = expf(lg[tid] - m) / s;
        int rank = 0;                               // jax top_k tie-break: lowest index
        #pragma unroll
        for (int j = 0; j < N_EXP; ++j)
            rank += (lg[j] > lg[tid]) || (lg[j] == lg[tid] && j < tid);
        cw[tok * N_EXP + tid] = (rank < K_TOP) ? g : 0.f;
    }
}

// ---------------- 2) capacity selection ------------------------------------------
__global__ __launch_bounds__(256) void capacity_kernel(
    const float* __restrict__ cw, float* __restrict__ fw)
{
    const int b = blockIdx.x / N_EXP;
    const int n = blockIdx.x % N_EXP;
    const int t = threadIdx.x;
    __shared__ float v[T_TOK];
    if (t < T_TOK) v[t] = cw[((size_t)(t * B_BAT + b)) * N_EXP + n];
    __syncthreads();
    if (t < T_TOK) {
        const float mv = v[t];
        int rank = 0;
        for (int tp = 0; tp < T_TOK; ++tp)
            rank += (v[tp] > mv) || (v[tp] == mv && tp < t);
        fw[((size_t)(t * B_BAT + b)) * N_EXP + n] = (rank < CAP) ? mv : 0.f;
    }
}

// ---------------- 3) deterministic per-expert token lists (one wave each) --------
__global__ __launch_bounds__(32) void build_lists_kernel(
    const float* __restrict__ fw, int* __restrict__ slot_tok, float* __restrict__ slot_w)
{
    const int e = blockIdx.x, lane = threadIdx.x;
    for (int i = lane; i < SLOTS; i += 32) {
        slot_tok[e * SLOTS + i] = -1;
        slot_w  [e * SLOTS + i] = 0.f;
    }
    int count = 0;
    for (int base = 0; base < TB; base += 32) {
        const int tok = base + lane;
        const float w = fw[(size_t)tok * N_EXP + e];
        const int pred = (w > 0.f);
        const unsigned long long mask = __ballot(pred);
        const int off = __popcll(mask & ((1ull << lane) - 1ull));
        if (pred) {
            const int s = count + off;
            if (s < SLOTS) { slot_tok[e * SLOTS + s] = tok; slot_w[e * SLOTS + s] = w; }
        }
        count += __popcll(mask);
    }
}

// ---------------- 4) out = -x ----------------------------------------------------
__global__ __launch_bounds__(256) void init_out_kernel(
    const float* __restrict__ x, float* __restrict__ out)
{
    const size_t i = (size_t)blockIdx.x * 256 + threadIdx.x;
    out[i] = -x[i];
}

// ------- 5) transpose + fp32->bf16 convert: in [R][C] fp32 -> out [C][R] bf16 ----
__global__ __launch_bounds__(256) void transpose_convert_kernel(
    const float* __restrict__ in, unsigned short* __restrict__ out, int R, int C)
{
    __shared__ float tile[64][65];
    const int r0 = blockIdx.x * 64, c0 = blockIdx.y * 64;
    const int tid = threadIdx.x;
    const int tr  = tid >> 4;          // 0..15
    const int tc4 = (tid & 15) * 4;    // 0..60
    #pragma unroll
    for (int j = 0; j < 4; ++j) {
        const int r = tr + j * 16;
        float4 f = *(const float4*)(in + (size_t)(r0 + r) * C + c0 + tc4);
        tile[r][tc4 + 0] = f.x; tile[r][tc4 + 1] = f.y;
        tile[r][tc4 + 2] = f.z; tile[r][tc4 + 3] = f.w;
    }
    __syncthreads();
    #pragma unroll
    for (int j = 0; j < 4; ++j) {
        const int c = tr + j * 16;     // output row = original column
        u16x4 h = { f2bf(tile[tc4 + 0][c]), f2bf(tile[tc4 + 1][c]),
                    f2bf(tile[tc4 + 2][c]), f2bf(tile[tc4 + 3][c]) };
        *(u16x4*)(out + (size_t)(c0 + c) * R + r0 + tc4) = h;
    }
}

// manual A-tile gather (one token row per thread), fp32 x -> bf16 LDS
__device__ __forceinline__ void stage_a_gather(
    unsigned short* dst, const float* __restrict__ x,
    const int* rowTok, int tid, int k0)
{
    const int tok = rowTok[tid];
    const float* src = x + (size_t)(tok < 0 ? 0 : tok) * F_DIM + k0;
    #pragma unroll
    for (int q = 0; q < 4; ++q) {
        float4 f0 = *(const float4*)(src + q * 8);
        float4 f1 = *(const float4*)(src + q * 8 + 4);
        if (tok < 0) { f0 = make_float4(0, 0, 0, 0); f1 = make_float4(0, 0, 0, 0); }
        u16x8 h = { f2bf(f0.x), f2bf(f0.y), f2bf(f0.z), f2bf(f0.w),
                    f2bf(f1.x), f2bf(f1.y), f2bf(f1.z), f2bf(f1.w) };
        *(u16x8*)(dst + tid * BK + q * 8) = h;
    }
}

// ---------------- 6a) Hh = relu( gather(x) * W1t[e]^T + b1[e] ) ------------------
// Wt: transposed bf16 weights [H][F]; B tile staged by TDM, A by manual gather.
__global__ __launch_bounds__(128) void gemm1_kernel(
    const float* __restrict__ x, const unsigned short* __restrict__ Wt,
    const float* __restrict__ b1, const int* __restrict__ slot_tok,
    unsigned short* __restrict__ Hh, int e)
{
    const int m0 = blockIdx.x * BM, n0 = blockIdx.y * BN;
    const int tid = threadIdx.x, wave = tid >> 5, lane = tid & 31;
    const int wm = wave >> 1, wn = wave & 1;

    __shared__ __align__(16) unsigned short As[2][BM * BK];
    __shared__ __align__(16) unsigned short Bs[2][BN * BK];
    __shared__ int   rowTok[BM];
    __shared__ float biasS[BN];

    rowTok[tid] = slot_tok[e * SLOTS + m0 + tid];
    biasS[tid]  = b1[e * H_DIM + n0 + tid];
    __syncthreads();

    v8f acc[4][4];
    #pragma unroll
    for (int i = 0; i < 4; ++i)
        #pragma unroll
        for (int j = 0; j < 4; ++j) acc[i][j] = (v8f)(0.f);

    // prologue: stage step 0
    stage_a_gather(&As[0][0], x, rowTok, tid, 0);
    if (tid < 32)
        tdm_load_2d(lds_offset(&Bs[0][0]), Wt + (size_t)n0 * F_DIM,
                    F_DIM, H_DIM, BK, BN, F_DIM);

    const int NSTEP = F_DIM / BK;   // 24
    for (int s = 0; s < NSTEP; ++s) {
        const int cur = s & 1;
        __syncthreads();                        // buf[cur^1] free (prev compute done)
        if (s + 1 < NSTEP) {
            stage_a_gather(&As[cur ^ 1][0], x, rowTok, tid, (s + 1) * BK);
            if (tid < 32) {
                tdm_load_2d(lds_offset(&Bs[cur ^ 1][0]),
                            Wt + (size_t)n0 * F_DIM + (size_t)(s + 1) * BK,
                            F_DIM, H_DIM, BK, BN, F_DIM);
                __builtin_amdgcn_s_wait_tensorcnt(1);   // step s's TDM finished
            }
        } else if (tid < 32) {
            __builtin_amdgcn_s_wait_tensorcnt(0);
        }
        __syncthreads();                        // publish buf[cur]
        mma_step(&As[cur][0], &Bs[cur][0], acc, wm, wn, lane);
    }

    // epilogue: relu(+bias) -> bf16 Hh
    const int half = (lane >> 4) << 3;
    #pragma unroll
    for (int i = 0; i < 4; ++i)
        #pragma unroll
        for (int j = 0; j < 4; ++j) {
            const int gcl = wn * 64 + j * 16 + (lane & 15);
            const int gc  = n0 + gcl;
            #pragma unroll
            for (int v = 0; v < 8; ++v) {
                const int rr = m0 + wm * 64 + i * 16 + v + half;
                const float val = fmaxf(acc[i][j][v] + biasS[gcl], 0.f);
                Hh[(size_t)rr * H_DIM + gc] = f2bf(val);
            }
        }
}

// ---------------- 6b) out[tok] += w_slot * ( Hh * W2t[e]^T + b2[e] ) -------------
// Both tiles staged by TDM (Hh and Wt are bf16 already). Wt: [F][H].
__global__ __launch_bounds__(128) void gemm2_kernel(
    const unsigned short* __restrict__ Hh, const unsigned short* __restrict__ Wt,
    const float* __restrict__ b2, const int* __restrict__ slot_tok,
    const float* __restrict__ slot_w, float* __restrict__ out, int e)
{
    const int m0 = blockIdx.x * BM, n0 = blockIdx.y * BN;
    const int tid = threadIdx.x, wave = tid >> 5, lane = tid & 31;
    const int wm = wave >> 1, wn = wave & 1;

    __shared__ __align__(16) unsigned short As[2][BM * BK];
    __shared__ __align__(16) unsigned short Bs[2][BN * BK];
    __shared__ int   rowTok[BM];
    __shared__ float rowW[BM];
    __shared__ float biasS[BN];

    rowTok[tid] = slot_tok[e * SLOTS + m0 + tid];
    rowW[tid]   = slot_w  [e * SLOTS + m0 + tid];
    biasS[tid]  = b2[e * F_DIM + n0 + tid];

    v8f acc[4][4];
    #pragma unroll
    for (int i = 0; i < 4; ++i)
        #pragma unroll
        for (int j = 0; j < 4; ++j) acc[i][j] = (v8f)(0.f);

    if (tid < 32) {
        tdm_load_2d(lds_offset(&As[0][0]), Hh + (size_t)m0 * H_DIM,
                    H_DIM, SLOTS, BK, BM, H_DIM);
        tdm_load_2d(lds_offset(&Bs[0][0]), Wt + (size_t)n0 * H_DIM,
                    H_DIM, F_DIM, BK, BN, H_DIM);
    }

    const int NSTEP = H_DIM / BK;   // 96
    for (int s = 0; s < NSTEP; ++s) {
        const int cur = s & 1;
        __syncthreads();
        if (tid < 32) {
            if (s + 1 < NSTEP) {
                const size_t k1 = (size_t)(s + 1) * BK;
                tdm_load_2d(lds_offset(&As[cur ^ 1][0]), Hh + (size_t)m0 * H_DIM + k1,
                            H_DIM, SLOTS, BK, BM, H_DIM);
                tdm_load_2d(lds_offset(&Bs[cur ^ 1][0]), Wt + (size_t)n0 * H_DIM + k1,
                            H_DIM, F_DIM, BK, BN, H_DIM);
                __builtin_amdgcn_s_wait_tensorcnt(2);   // only newest 2 outstanding
            } else {
                __builtin_amdgcn_s_wait_tensorcnt(0);
            }
        }
        __syncthreads();
        mma_step(&As[cur][0], &Bs[cur][0], acc, wm, wn, lane);
    }

    // epilogue: out[tok, gc] += w * (acc + b2)  (slots unique; experts serialized)
    const int half = (lane >> 4) << 3;
    #pragma unroll
    for (int i = 0; i < 4; ++i)
        #pragma unroll
        for (int j = 0; j < 4; ++j) {
            const int gcl = wn * 64 + j * 16 + (lane & 15);
            const int gc  = n0 + gcl;
            #pragma unroll
            for (int v = 0; v < 8; ++v) {
                const int rloc = wm * 64 + i * 16 + v + half;
                const int tok  = rowTok[rloc];
                if (tok >= 0) {
                    const float val = rowW[rloc] * (acc[i][j][v] + biasS[gcl]);
                    float* p = out + (size_t)tok * F_DIM + gc;
                    *p = *p + val;
                }
            }
        }
}

// ---------------- host-side orchestration ----------------------------------------
extern "C" void kernel_launch(void* const* d_in, const int* in_sizes, int n_in,
                              void* d_out, int out_size, void* d_ws, size_t ws_size,
                              hipStream_t stream) {
    const float* x   = (const float*)d_in[0];
    const float* Wg1 = (const float*)d_in[1];
    const float* bg1 = (const float*)d_in[2];
    const float* Wg2 = (const float*)d_in[3];
    const float* bg2 = (const float*)d_in[4];
    const float* W1  = (const float*)d_in[5];
    const float* b1  = (const float*)d_in[6];
    const float* W2  = (const float*)d_in[7];
    const float* b2  = (const float*)d_in[8];
    float* out = (float*)d_out;

    size_t off = 0;
    auto bump = [&](size_t bytes) -> void* {
        void* p = (char*)d_ws + off;
        off += (bytes + 255) & ~(size_t)255;
        return p;
    };
    float*          cw       = (float*)bump((size_t)TB * N_EXP * sizeof(float));
    float*          fw       = (float*)bump((size_t)TB * N_EXP * sizeof(float));
    int*            slot_tok = (int*)  bump((size_t)N_EXP * SLOTS * sizeof(int));
    float*          slot_w   = (float*)bump((size_t)N_EXP * SLOTS * sizeof(float));
    unsigned short* Hh       = (unsigned short*)bump((size_t)SLOTS * H_DIM * sizeof(unsigned short));
    unsigned short* W1t      = (unsigned short*)bump((size_t)H_DIM * F_DIM * sizeof(unsigned short));
    unsigned short* W2t      = (unsigned short*)bump((size_t)F_DIM * H_DIM * sizeof(unsigned short));
    (void)ws_size; (void)in_sizes; (void)n_in; (void)out_size;

    gate_kernel       <<<TB,            256, 0, stream>>>(x, Wg1, bg1, Wg2, bg2, cw);
    capacity_kernel   <<<B_BAT * N_EXP, 256, 0, stream>>>(cw, fw);
    build_lists_kernel<<<N_EXP,          32, 0, stream>>>(fw, slot_tok, slot_w);
    init_out_kernel   <<<(TB * F_DIM) / 256, 256, 0, stream>>>(x, out);

    for (int e = 0; e < N_EXP; ++e) {
        // W1[e]: [F][H] fp32  ->  W1t: [H][F] bf16
        transpose_convert_kernel<<<dim3(F_DIM / 64, H_DIM / 64), 256, 0, stream>>>(
            W1 + (size_t)e * F_DIM * H_DIM, W1t, F_DIM, H_DIM);
        gemm1_kernel<<<dim3(SLOTS / BM, H_DIM / BN), 128, 0, stream>>>(
            x, W1t, b1, slot_tok, Hh, e);
        // W2[e]: [H][F] fp32  ->  W2t: [F][H] bf16
        transpose_convert_kernel<<<dim3(H_DIM / 64, F_DIM / 64), 256, 0, stream>>>(
            W2 + (size_t)e * H_DIM * F_DIM, W2t, H_DIM, F_DIM);
        gemm2_kernel<<<dim3(SLOTS / BM, F_DIM / BN), 128, 0, stream>>>(
            Hh, W2t, b2, slot_tok, slot_w, out, e);
    }
}
